// AttentionFusion_11330123727429
// MI455X (gfx1250) — matrix operand order repaired and checked by reading, not compiled, not run on hardware
//
#include <hip/hip_runtime.h>
#include <hip/hip_bf16.h>

// ---------------------------------------------------------------------------
// AttentionFusion on gfx1250 (MI455X), bf16 WMMA pipeline:
//   K0: convert all weights fp32 -> bf16 into workspace (one-shot, 1.65 MB)
//   K1 (fused): stage embeddings -> projections -> QKV -> 4-head attention
//               (wave32 lane == head_dim) -> out-proj + residual + LayerNorm
//               -> y bf16 [B,640] in workspace
//   K2: fusion GEMM out[B,992] = y[B,640] @ fus_w^T + fus_b
//       64x128 WG tiles, B-fragment reuse across 4 M-tiles
// ---------------------------------------------------------------------------

typedef __attribute__((ext_vector_type(16))) __bf16 v16bf;
typedef __attribute__((ext_vector_type(8)))  float  v8f;

#define WMMA_BF16(a, b, c) \
  __builtin_amdgcn_wmma_f32_16x16x32_bf16(false, (a), false, (b), (short)0, (c), false, false)

// --- fragment builders per CDNA5 ISA 7.12.2 (wave32) -----------------------
// A (16x32 bf16): lane l -> row M = l&15 ; kh = (l>>4)*8 ;
//   elements 0..7  = K in [kh, kh+8) ; elements 8..15 = K in [kh+16, kh+24)
__device__ __forceinline__ v16bf a_frag_bf16(const __bf16* A, int ld,
                                             int row_base, int k0) {
  int l  = threadIdx.x & 31;
  const __bf16* p = A + (size_t)(row_base + (l & 15)) * ld + k0 + (l >> 4) * 8;
  v16bf out;
#pragma unroll
  for (int i = 0; i < 8; ++i) out[i] = p[i];
#pragma unroll
  for (int i = 0; i < 8; ++i) out[8 + i] = p[16 + i];
  return out;
}

// B (32x16 bf16): B[k][n] = W[n][k], W row-major [N,K] bf16.
// lane l -> col N = n_base + (l&15) ; kh = (l>>4)*8 ; elements 0..7 = K [kh,kh+8), 8..15 = K [kh+16,kh+24)
__device__ __forceinline__ v16bf b_frag_bf16(const __bf16* W, int ldk,
                                             int n_base, int k0) {
  int l = threadIdx.x & 31;
  const __bf16* p = W + (size_t)(n_base + (l & 15)) * ldk + k0 + (l >> 4) * 8;
  v16bf out;
#pragma unroll
  for (int i = 0; i < 8; ++i) out[i] = p[i];
#pragma unroll
  for (int i = 0; i < 8; ++i) out[8 + i] = p[16 + i];
  return out;
}

// C/D (16x16 f32): lane l -> col N = l&15 ; VGPR r -> row M = (l>>4)*8 + r

// ---------------------------------------------------------------------------
// K0: weight conversion fp32 -> bf16 (segment layout, element offsets):
//   Wc 0 | Wm 16384 | Wd 49152 | Wp 57344 | Wr 61440 | in_w 126976
//   out_w 176128 | fus_w 192512 | total 827392
// ---------------------------------------------------------------------------
#define SEG_WC    0
#define SEG_WM    16384
#define SEG_WD    49152
#define SEG_WP    57344
#define SEG_WR    61440
#define SEG_INW   126976
#define SEG_OUTW  176128
#define SEG_FUSW  192512
#define SEG_TOTAL 827392

__global__ __launch_bounds__(256)
void AF_convert_weights(const float* __restrict__ Wc, const float* __restrict__ Wm,
                        const float* __restrict__ Wd, const float* __restrict__ Wp,
                        const float* __restrict__ Wr, const float* __restrict__ in_w,
                        const float* __restrict__ out_w, const float* __restrict__ fus_w,
                        __bf16* __restrict__ dst) {
  int idx = blockIdx.x * 256 + threadIdx.x;
  if (idx >= SEG_TOTAL) return;
  const float* src; int off;
  if      (idx < SEG_WM)   { src = Wc;    off = idx - SEG_WC; }
  else if (idx < SEG_WD)   { src = Wm;    off = idx - SEG_WM; }
  else if (idx < SEG_WP)   { src = Wd;    off = idx - SEG_WD; }
  else if (idx < SEG_WR)   { src = Wp;    off = idx - SEG_WP; }
  else if (idx < SEG_INW)  { src = Wr;    off = idx - SEG_WR; }
  else if (idx < SEG_OUTW) { src = in_w;  off = idx - SEG_INW; }
  else if (idx < SEG_FUSW) { src = out_w; off = idx - SEG_OUTW; }
  else                     { src = fus_w; off = idx - SEG_FUSW; }
  dst[idx] = (__bf16)src[off];
}

// ---------------------------------------------------------------------------
// K1: 8 samples / workgroup, 256 threads (8 waves).
// token rows = sample*5 + modality; 40 valid, all buffers padded to 48 rows
// so pad rows are in-bounds garbage feeding only discarded outputs (no guards).
// LDS: xb bf16[48][128] @0 (12288) | qkvb bf16[48][384] @12288 (36864,
//      aliased by ebuf bf16[7936+] during staging and yacc f32[48][128] after
//      attention) | ctxb bf16[48][128] @49152 (12288). Total 61440 B.
// ---------------------------------------------------------------------------
#define SMEM_XB     0
#define SMEM_QKV    12288
#define SMEM_CTX    49152
#define SMEM_TOTAL  61440

__global__ __launch_bounds__(256)
void AF_proj_attn_ln_kernel(const float* __restrict__ cat,  const float* __restrict__ menu,
                            const float* __restrict__ din,  const float* __restrict__ prc,
                            const float* __restrict__ rev,
                            const __bf16* __restrict__ wb,   // converted weights
                            const float* __restrict__ bc, const float* __restrict__ bm,
                            const float* __restrict__ bd, const float* __restrict__ bp,
                            const float* __restrict__ br,
                            const float* __restrict__ in_b, const float* __restrict__ out_b,
                            const float* __restrict__ ln_g, const float* __restrict__ ln_b,
                            __bf16* __restrict__ y_out) {
  __shared__ __align__(16) unsigned char smem[SMEM_TOTAL];
  __bf16* xb   = (__bf16*)(smem + SMEM_XB);
  __bf16* qkvb = (__bf16*)(smem + SMEM_QKV);
  __bf16* ebuf = (__bf16*)(smem + SMEM_QKV);   // staging alias (pre-QKV)
  __bf16* ctxb = (__bf16*)(smem + SMEM_CTX);
  float*  yacc = (float*)(smem + SMEM_QKV);    // alias (post-attention)

  const int w    = threadIdx.x >> 5;           // wave 0..7
  const int lane = threadIdx.x & 31;
  const int s0   = blockIdx.x * 8;             // first sample of this WG

  const float*  embs[5]  = {cat, menu, din, prc, rev};
  const float*  bias5[5] = {bc, bm, bd, bp, br};
  const int     Ks[5]    = {128, 256, 64, 32, 512};
  const int     moff[5]  = {0, 1024, 3072, 3584, 3840};  // ebuf segment offsets
  const __bf16* Wseg[5]  = {wb + SEG_WC, wb + SEG_WM, wb + SEG_WD,
                            wb + SEG_WP, wb + SEG_WR};

  // ---- stage 0: cooperatively stage 8 samples' embeddings as bf16 in LDS
#pragma unroll
  for (int m = 0; m < 5; ++m) {
    int n = 8 * Ks[m];
    for (int i = threadIdx.x; i < n; i += 256) {
      int s = i / Ks[m];                       // power-of-two -> shift
      int c = i - s * Ks[m];
      ebuf[moff[m] + i] = (__bf16)embs[m][(size_t)(s0 + s) * Ks[m] + c];
    }
  }
  __syncthreads();

  // ---- stage 1: projections; wave w owns cols [w*16, w*16+16) of ATT=128
#pragma unroll
  for (int m = 0; m < 5; ++m) {
    v8f acc = {};
    for (int k0 = 0; k0 < Ks[m]; k0 += 32) {
      v16bf a = a_frag_bf16(ebuf + moff[m], Ks[m], 0, k0);  // rows 8..15 garbage
      v16bf b = b_frag_bf16(Wseg[m], Ks[m], w * 16, k0);
      acc = WMMA_BF16(a, b, acc);
    }
    if (lane < 16) {                           // lanes 0..15 carry M = 0..7 (valid)
      int n = w * 16 + lane;
      float bv = bias5[m][n];
#pragma unroll
      for (int r = 0; r < 8; ++r)
        xb[(r * 5 + m) * 128 + n] = (__bf16)(acc[r] + bv);
    }
  }
  // pad rows 40..47 of xb left as stale data; they feed only discarded rows
  __syncthreads();

  // ---- stage 2: QKV = x @ in_w^T + in_b  (M=48, N=384, K=128)
  const __bf16* in_wb = wb + SEG_INW;
  for (int t = w; t < 3 * 24; t += 8) {
    int mt = t / 24, nt = t % 24;
    v8f acc = {};
#pragma unroll
    for (int k0 = 0; k0 < 128; k0 += 32) {
      v16bf a = a_frag_bf16(xb, 128, mt * 16, k0);
      v16bf b = b_frag_bf16(in_wb, 128, nt * 16, k0);
      acc = WMMA_BF16(a, b, acc);
    }
    int n  = nt * 16 + (lane & 15);
    int rb = mt * 16 + ((lane >> 4) << 3);
    float bv = in_b[n];
#pragma unroll
    for (int r = 0; r < 8; ++r)
      qkvb[(rb + r) * 384 + n] = (__bf16)(acc[r] + bv);   // rows 40..47 harmless
  }
  __syncthreads();

  // ---- stage 3: attention; wave w == local sample w, lane == head-dim (DH=32)
#pragma unroll
  for (int h = 0; h < 4; ++h) {
    float q[5], kk[5], vv[5];
#pragma unroll
    for (int i = 0; i < 5; ++i) {
      const __bf16* row = qkvb + (w * 5 + i) * 384;
      q[i]  = (float)row[h * 32 + lane];
      kk[i] = (float)row[128 + h * 32 + lane];
      vv[i] = (float)row[256 + h * 32 + lane];
    }
    float sc[5][5];
#pragma unroll
    for (int i = 0; i < 5; ++i)
#pragma unroll
      for (int j = 0; j < 5; ++j) {
        float p = q[i] * kk[j];
        p += __shfl_xor(p, 16, 32);
        p += __shfl_xor(p, 8, 32);
        p += __shfl_xor(p, 4, 32);
        p += __shfl_xor(p, 2, 32);
        p += __shfl_xor(p, 1, 32);
        sc[i][j] = p * 0.17677669529663687f;   // 1/sqrt(32)
      }
#pragma unroll
    for (int i = 0; i < 5; ++i) {
      float mx = sc[i][0];
#pragma unroll
      for (int j = 1; j < 5; ++j) mx = fmaxf(mx, sc[i][j]);
      float s = 0.f;
#pragma unroll
      for (int j = 0; j < 5; ++j) { sc[i][j] = __expf(sc[i][j] - mx); s += sc[i][j]; }
      float inv = 1.0f / s;
#pragma unroll
      for (int j = 0; j < 5; ++j) sc[i][j] *= inv;
    }
#pragma unroll
    for (int i = 0; i < 5; ++i) {
      float c = 0.f;
#pragma unroll
      for (int j = 0; j < 5; ++j) c += sc[i][j] * vv[j];
      ctxb[(w * 5 + i) * 128 + h * 32 + lane] = (__bf16)c;
    }
  }
  __syncthreads();

  // ---- stage 4: out-proj + residual  (M=48, N=128, K=128) -> yacc f32
  const __bf16* out_wb = wb + SEG_OUTW;
  for (int t = w; t < 3 * 8; t += 8) {
    int mt = t / 8, nt = t % 8;
    v8f acc = {};
#pragma unroll
    for (int k0 = 0; k0 < 128; k0 += 32) {
      v16bf a = a_frag_bf16(ctxb, 128, mt * 16, k0);
      v16bf b = b_frag_bf16(out_wb, 128, nt * 16, k0);
      acc = WMMA_BF16(a, b, acc);
    }
    int n  = nt * 16 + (lane & 15);
    int rb = mt * 16 + ((lane >> 4) << 3);
    float bv = out_b[n];
#pragma unroll
    for (int r = 0; r < 8; ++r) {
      int row = rb + r;
      yacc[row * 128 + n] = acc[r] + bv + (float)xb[row * 128 + n];  // rows >=40 harmless
    }
  }
  __syncthreads();

  // ---- stage 5: LayerNorm over 128 cols; wave w handles rows w*5 .. w*5+4
#pragma unroll
  for (int i = 0; i < 5; ++i) {
    int row = w * 5 + i;
    float vals[4];
    float s = 0.f, s2 = 0.f;
#pragma unroll
    for (int c = 0; c < 4; ++c) {
      vals[c] = yacc[row * 128 + c * 32 + lane];
      s  += vals[c];
      s2 += vals[c] * vals[c];
    }
#pragma unroll
    for (int mm = 16; mm >= 1; mm >>= 1) {
      s  += __shfl_xor(s,  mm, 32);
      s2 += __shfl_xor(s2, mm, 32);
    }
    float mu  = s * (1.0f / 128.0f);
    float var = s2 * (1.0f / 128.0f) - mu * mu;
    float inv = rsqrtf(var + 1e-5f);
    int sg = s0 + w;                            // global sample; modality = i
#pragma unroll
    for (int c = 0; c < 4; ++c) {
      int col = c * 32 + lane;
      float yv = (vals[c] - mu) * inv * ln_g[col] + ln_b[col];
      y_out[(size_t)sg * 640 + i * 128 + col] = (__bf16)yv;
    }
  }
}

// ---------------------------------------------------------------------------
// K2: fusion GEMM  out[B,992] = y[B,640] @ fus_w^T + fus_b
// WG tile = 64 rows x 128 cols; wave -> one N-tile, 4 M-tiles (B-frag reuse 4x).
// A chunk [64][32] double-role staged in LDS per k-step (4 KB).
// ---------------------------------------------------------------------------
__global__ __launch_bounds__(256)
void AF_fusion_kernel(const __bf16* __restrict__ Y, const __bf16* __restrict__ Wf,
                      const float* __restrict__ bias, float* __restrict__ out) {
  __shared__ __align__(16) __bf16 at[64 * 32];     // 4 KB A chunk
  const int row0 = blockIdx.x * 64;
  const int w    = threadIdx.x >> 5;
  const int lane = threadIdx.x & 31;
  const int nt   = blockIdx.y * 8 + w;             // 992/16 = 62 tiles; >=62 idle

  v8f acc[4] = {{}, {}, {}, {}};
  for (int k0 = 0; k0 < 640; k0 += 32) {
    __syncthreads();                                // protect previous-iter reads
    for (int i = threadIdx.x; i < 64 * 32; i += 256) {
      int r = i >> 5, c = i & 31;
      at[i] = Y[(size_t)(row0 + r) * 640 + k0 + c];
    }
    __syncthreads();
    if (nt < 62) {
      __builtin_prefetch(Wf + (size_t)(nt * 16 + (lane & 15)) * 640 + k0 + 32, 0, 1);
      v16bf b = b_frag_bf16(Wf, 640, nt * 16, k0);
#pragma unroll
      for (int mt = 0; mt < 4; ++mt) {
        v16bf a = a_frag_bf16(at, 32, mt * 16, 0);
        acc[mt] = WMMA_BF16(a, b, acc[mt]);
      }
    }
  }
  if (nt >= 62) return;

  int n  = nt * 16 + (lane & 15);
  int rb = (lane >> 4) << 3;
  float bv = bias[n];
#pragma unroll
  for (int mt = 0; mt < 4; ++mt)
#pragma unroll
    for (int r = 0; r < 8; ++r)
      out[(size_t)(row0 + mt * 16 + rb + r) * 992 + n] = acc[mt][r] + bv;
}

// ---------------------------------------------------------------------------
extern "C" void kernel_launch(void* const* d_in, const int* in_sizes, int n_in,
                              void* d_out, int out_size, void* d_ws, size_t ws_size,
                              hipStream_t stream) {
  const float* cat  = (const float*)d_in[0];
  const float* menu = (const float*)d_in[1];
  const float* din  = (const float*)d_in[2];
  const float* prc  = (const float*)d_in[3];
  const float* rev  = (const float*)d_in[4];
  const float* Wc = (const float*)d_in[5];  const float* bc = (const float*)d_in[6];
  const float* Wm = (const float*)d_in[7];  const float* bm = (const float*)d_in[8];
  const float* Wd = (const float*)d_in[9];  const float* bd = (const float*)d_in[10];
  const float* Wp = (const float*)d_in[11]; const float* bp = (const float*)d_in[12];
  const float* Wr = (const float*)d_in[13]; const float* br = (const float*)d_in[14];
  const float* in_w  = (const float*)d_in[15]; const float* in_b  = (const float*)d_in[16];
  const float* out_w = (const float*)d_in[17]; const float* out_b = (const float*)d_in[18];
  const float* ln_g  = (const float*)d_in[19]; const float* ln_b  = (const float*)d_in[20];
  const float* fus_w = (const float*)d_in[21]; const float* fus_b = (const float*)d_in[22];

  const int B = in_sizes[0] / 128;                 // 65536
  __bf16* y_ws = (__bf16*)d_ws;                    // B*640 bf16 = 84 MB
  __bf16* wb   = (__bf16*)((char*)d_ws + (size_t)B * 640 * sizeof(__bf16));
  float*  out  = (float*)d_out;                    // B*992 f32

  AF_convert_weights<<<dim3((SEG_TOTAL + 255) / 256), dim3(256), 0, stream>>>(
      Wc, Wm, Wd, Wp, Wr, in_w, out_w, fus_w, wb);

  AF_proj_attn_ln_kernel<<<dim3(B / 8), dim3(256), 0, stream>>>(
      cat, menu, din, prc, rev, wb,
      bc, bm, bd, bp, br, in_b, out_b, ln_g, ln_b, y_ws);

  AF_fusion_kernel<<<dim3(B / 64, 8), dim3(256), 0, stream>>>(
      y_ws, wb + SEG_FUSW, fus_b, out);
}
